// MoELayer_42640435314740
// MI455X (gfx1250) — matrix operand order repaired and checked by reading, not compiled, not run on hardware
//
#include <hip/hip_runtime.h>
#include <hip/hip_bf16.h>

// ---------------------------------------------------------------------------
// MoE top-2 dispatch for MI455X (gfx1250, wave32, WMMA bf16 16x16x32)
// ---------------------------------------------------------------------------

typedef __attribute__((ext_vector_type(16))) __bf16 v16bf;
typedef __attribute__((ext_vector_type(8)))  __bf16 v8bf;
typedef __attribute__((ext_vector_type(8)))  float  v8f;

#define NTOK 8192   // B*T = 4*2048
#define DDIM 512
#define FDIM 1024
#define NEXP 8

static __device__ inline v8f wmma_bf16(v16bf a, v16bf b, v8f c) {
  // D = A(16x32 bf16) * B(32x16 bf16) + C(16x16 f32)
  return __builtin_amdgcn_wmma_f32_16x16x32_bf16(
      /*neg_a=*/false, a, /*neg_b=*/false, b,
      /*c_mod=*/(short)0, c, /*reuse_a=*/false, /*reuse_b=*/false);
}

// B fragment: lane l holds B[k0 + (l/16)*16 + j][n = nTile + l%16], j = 0..15
// contiguous in our K-major transposed weight layout -> one 32B load.
static __device__ inline v16bf load_b16(const __bf16* p) {
  return *(const v16bf*)p;   // 32B aligned by construction
}

// A fragment from a bf16 row: a[j] = row[k0 + half*16 + j]
static __device__ inline v16bf load_a_bf16(const __bf16* row, int k0, int kh) {
  v16bf a;
  ((v8bf*)&a)[0] = *(const v8bf*)(row + k0 + kh * 16);
  ((v8bf*)&a)[1] = *(const v8bf*)(row + k0 + kh * 16 + 8);
  return a;
}

// ---------------------------------------------------------------------------
// 0) zero output + counters (output is accumulated with atomics later)
// ---------------------------------------------------------------------------
__global__ void zero_init(float* __restrict__ out, int* __restrict__ cnt,
                          int* __restrict__ fill) {
  const size_t i = (size_t)blockIdx.x * blockDim.x + threadIdx.x;
  out[i] = 0.f;                               // grid sized exactly NTOK*DDIM
  if (i < NEXP) { cnt[i] = 0; fill[i] = 0; }
}

// ---------------------------------------------------------------------------
// 1a) activations fp32 -> bf16 once (removes cvt + half the A-bytes from the
//     hot GEMM loops; x is re-read ~32x across f-tiles and top-2 slots)
// ---------------------------------------------------------------------------
__global__ void cvt_x(const float* __restrict__ in, __bf16* __restrict__ out) {
  const size_t i = ((size_t)blockIdx.x * blockDim.x + threadIdx.x) * 4;
  const float4 v = *(const float4*)(in + i);
  __bf16* o = out + i;
  o[0] = (__bf16)v.x; o[1] = (__bf16)v.y; o[2] = (__bf16)v.z; o[3] = (__bf16)v.w;
}

// ---------------------------------------------------------------------------
// 1b) weight fp32 [E][K][N] -> bf16 [E][N][K]  (K contiguous for B fragments)
// ---------------------------------------------------------------------------
__global__ void transpose_cvt(const float* __restrict__ in,
                              __bf16* __restrict__ out, int K, int N) {
  __shared__ float t[32][33];
  const int e = blockIdx.z;
  const float* src = in + (size_t)e * K * N;
  __bf16* dst = out + (size_t)e * N * K;
  const int n0 = blockIdx.x * 32, k0 = blockIdx.y * 32;
  const int tx = threadIdx.x, ty = threadIdx.y;
#pragma unroll
  for (int j = 0; j < 4; ++j)
    t[ty + j * 8][tx] = src[(size_t)(k0 + ty + j * 8) * N + n0 + tx];
  __syncthreads();
#pragma unroll
  for (int j = 0; j < 4; ++j)
    dst[(size_t)(n0 + ty + j * 8) * K + k0 + tx] = (__bf16)t[tx][ty + j * 8];
}

// ---------------------------------------------------------------------------
// 2) router: one wave per token. logits -> top2 -> renormalized weights.
//    (softmax then top-k renorm == softmax over the two top logits)
// ---------------------------------------------------------------------------
__global__ void __launch_bounds__(256)
router_topk(const float* __restrict__ x, const float* __restrict__ rw,
            const float* __restrict__ rb, int* __restrict__ topIdx,
            float* __restrict__ topW, int* __restrict__ cnt) {
  const int t = blockIdx.x * 8 + (threadIdx.x >> 5);
  const int lane = threadIdx.x & 31;
  const float* xr = x + (size_t)t * DDIM;
  float acc[NEXP];
#pragma unroll
  for (int e = 0; e < NEXP; ++e) acc[e] = 0.f;
  for (int k = lane; k < DDIM; k += 32) {
    const float xv = xr[k];
#pragma unroll
    for (int e = 0; e < NEXP; ++e) acc[e] += xv * rw[e * DDIM + k];
  }
#pragma unroll
  for (int e = 0; e < NEXP; ++e)
    for (int o = 16; o > 0; o >>= 1) acc[e] += __shfl_xor(acc[e], o, 32);
  if (lane == 0) {
    float lg[NEXP];
#pragma unroll
    for (int e = 0; e < NEXP; ++e) lg[e] = acc[e] + rb[e];
    int i0 = 0; float l0 = lg[0];
#pragma unroll
    for (int e = 1; e < NEXP; ++e) if (lg[e] > l0) { l0 = lg[e]; i0 = e; }
    int i1 = -1; float l1 = -3.4e38f;
#pragma unroll
    for (int e = 0; e < NEXP; ++e)
      if (e != i0 && lg[e] > l1) { l1 = lg[e]; i1 = e; }
    const float r  = __expf(l1 - l0);        // <= 1, numerically stable
    const float w0 = 1.f / (1.f + r);
    topIdx[2 * t]     = i0; topW[2 * t]     = w0;
    topIdx[2 * t + 1] = i1; topW[2 * t + 1] = 1.f - w0;
    atomicAdd(&cnt[i0], 1);
    atomicAdd(&cnt[i1], 1);
  }
}

// 3) 8-entry exclusive scan -> compact per-expert slot offsets
__global__ void scan8(const int* __restrict__ cnt, int* __restrict__ off) {
  if (threadIdx.x == 0) {
    int s = 0;
#pragma unroll
    for (int e = 0; e < NEXP; ++e) { off[e] = s; s += cnt[e]; }
    off[NEXP] = s;   // == 2*NTOK
  }
}

// 4) scatter (token, weight) into contiguous per-expert buckets
__global__ void scatter_slots(const int* __restrict__ topIdx,
                              const float* __restrict__ topW,
                              const int* __restrict__ off, int* __restrict__ fill,
                              int* __restrict__ btok, float* __restrict__ bw) {
  const int s = blockIdx.x * blockDim.x + threadIdx.x;
  if (s >= 2 * NTOK) return;
  const int e = topIdx[s];
  const int row = off[e] + atomicAdd(&fill[e], 1);
  btok[row] = s >> 1;
  bw[row]   = topW[s];
}

// ---------------------------------------------------------------------------
// 5) grouped GEMM 1: H[slot, f] = silu(x@Wg) * (x@Wu), bf16 out.
//    block tile 64 rows x 64 f-cols, 8 waves = 4M x 2N, K = 512.
//    Double-buffered fragments: loads for step k+1 issue before WMMAs of k.
// ---------------------------------------------------------------------------
__global__ void __launch_bounds__(256)
moe_gateup(const __bf16* __restrict__ xbf, const __bf16* __restrict__ wgu_t,
           const int* __restrict__ btok, const int* __restrict__ off,
           __bf16* __restrict__ H) {
  const int e = blockIdx.z;
  const int rowBase = off[e] + blockIdx.y * 64;
  const int rows = off[e + 1] - rowBase;
  if (rows <= 0) return;                       // block-uniform exit
  const int fBase = blockIdx.x * 64;
  const int wave = threadIdx.x >> 5, lane = threadIdx.x & 31;
  const int wm = wave >> 1, wn = wave & 1;
  const int kh = lane >> 4, l16 = lane & 15;

  const int m = wm * 16 + l16;                 // A-fragment row (both halves)
  const __bf16* xr = xbf + (size_t)btok[rowBase + (m < rows ? m : 0)] * DDIM;

  const __bf16* bg0 = wgu_t + ((size_t)e * 2 * FDIM + fBase + wn * 32 + l16) * DDIM;
  const __bf16* bg1 = bg0 + (size_t)16 * DDIM;
  const __bf16* bu0 = bg0 + (size_t)FDIM * DDIM;   // up = columns F..2F-1
  const __bf16* bu1 = bu0 + (size_t)16 * DDIM;

  v8f g0 = {}, g1 = {}, u0 = {}, u1 = {};
  // prologue: stage K-step 0
  v16bf a  = load_a_bf16(xr, 0, kh);
  v16bf b0 = load_b16(bg0 + kh * 16);
  v16bf b1 = load_b16(bg1 + kh * 16);
  v16bf b2 = load_b16(bu0 + kh * 16);
  v16bf b3 = load_b16(bu1 + kh * 16);
  for (int k0 = 0; k0 < DDIM - 32; k0 += 32) {
    const int kb = k0 + 32 + kh * 16;
    const v16bf an  = load_a_bf16(xr, k0 + 32, kh);   // prefetch step k+1
    const v16bf nb0 = load_b16(bg0 + kb);
    const v16bf nb1 = load_b16(bg1 + kb);
    const v16bf nb2 = load_b16(bu0 + kb);
    const v16bf nb3 = load_b16(bu1 + kb);
    g0 = wmma_bf16(a, b0, g0);
    g1 = wmma_bf16(a, b1, g1);
    u0 = wmma_bf16(a, b2, u0);
    u1 = wmma_bf16(a, b3, u1);
    a = an; b0 = nb0; b1 = nb1; b2 = nb2; b3 = nb3;
  }
  g0 = wmma_bf16(a, b0, g0);                   // peeled last step
  g1 = wmma_bf16(a, b1, g1);
  u0 = wmma_bf16(a, b2, u0);
  u1 = wmma_bf16(a, b3, u1);

#pragma unroll
  for (int r = 0; r < 8; ++r) {
    const int mrow = wm * 16 + kh * 8 + r;     // C layout: m = r + 8*(lane/16)
    if (mrow < rows) {
      __bf16* hrow = H + (size_t)(rowBase + mrow) * FDIM + fBase + wn * 32 + l16;
      float g = g0[r], u = u0[r];
      hrow[0]  = (__bf16)((g / (1.f + __expf(-g))) * u);
      g = g1[r]; u = u1[r];
      hrow[16] = (__bf16)((g / (1.f + __expf(-g))) * u);
    }
  }
}

// ---------------------------------------------------------------------------
// 6) grouped GEMM 2: out[token] += w * (H[slot] @ Wd).
//    block tile 64 rows x 128 d-cols, 8 waves = 4M x 2N, K = 1024.
// ---------------------------------------------------------------------------
__global__ void __launch_bounds__(256)
moe_down(const __bf16* __restrict__ H, const __bf16* __restrict__ wd_t,
         const int* __restrict__ btok, const float* __restrict__ bw,
         const int* __restrict__ off, float* __restrict__ out) {
  const int e = blockIdx.z;
  const int rowBase = off[e] + blockIdx.y * 64;
  const int rows = off[e + 1] - rowBase;
  if (rows <= 0) return;
  const int dBase = blockIdx.x * 128;
  const int wave = threadIdx.x >> 5, lane = threadIdx.x & 31;
  const int wm = wave >> 1, wn = wave & 1;
  const int kh = lane >> 4, l16 = lane & 15;

  const int m = wm * 16 + l16;
  const __bf16* hr = H + (size_t)(rowBase + (m < rows ? m : 0)) * FDIM;

  const __bf16* p0 = wd_t + ((size_t)e * DDIM + dBase + wn * 64 + l16) * FDIM;
  const __bf16* p1 = p0 + (size_t)16 * FDIM;
  const __bf16* p2 = p0 + (size_t)32 * FDIM;
  const __bf16* p3 = p0 + (size_t)48 * FDIM;

  v8f c0 = {}, c1 = {}, c2 = {}, c3 = {};
  v16bf a  = load_a_bf16(hr, 0, kh);
  v16bf b0 = load_b16(p0 + kh * 16);
  v16bf b1 = load_b16(p1 + kh * 16);
  v16bf b2 = load_b16(p2 + kh * 16);
  v16bf b3 = load_b16(p3 + kh * 16);
  for (int k0 = 0; k0 < FDIM - 32; k0 += 32) {
    const int kb = k0 + 32 + kh * 16;
    const v16bf an  = load_a_bf16(hr, k0 + 32, kh);   // prefetch step k+1
    const v16bf nb0 = load_b16(p0 + kb);
    const v16bf nb1 = load_b16(p1 + kb);
    const v16bf nb2 = load_b16(p2 + kb);
    const v16bf nb3 = load_b16(p3 + kb);
    c0 = wmma_bf16(a, b0, c0);
    c1 = wmma_bf16(a, b1, c1);
    c2 = wmma_bf16(a, b2, c2);
    c3 = wmma_bf16(a, b3, c3);
    a = an; b0 = nb0; b1 = nb1; b2 = nb2; b3 = nb3;
  }
  c0 = wmma_bf16(a, b0, c0);                   // peeled last step
  c1 = wmma_bf16(a, b1, c1);
  c2 = wmma_bf16(a, b2, c2);
  c3 = wmma_bf16(a, b3, c3);

#pragma unroll
  for (int r = 0; r < 8; ++r) {
    const int mrow = wm * 16 + kh * 8 + r;
    if (mrow < rows) {
      const int row = rowBase + mrow;
      const float w = bw[row];
      float* orow = out + (size_t)btok[row] * DDIM + dBase + wn * 64 + l16;
      atomicAdd(orow +  0, w * c0[r]);    // 2 commutative adds per element
      atomicAdd(orow + 16, w * c1[r]);
      atomicAdd(orow + 32, w * c2[r]);
      atomicAdd(orow + 48, w * c3[r]);
    }
  }
}

// ---------------------------------------------------------------------------
extern "C" void kernel_launch(void* const* d_in, const int* in_sizes, int n_in,
                              void* d_out, int out_size, void* d_ws, size_t ws_size,
                              hipStream_t stream) {
  (void)in_sizes; (void)n_in; (void)out_size; (void)ws_size;
  const float* x   = (const float*)d_in[0];
  const float* rw  = (const float*)d_in[1];
  const float* rb  = (const float*)d_in[2];
  const float* wgu = (const float*)d_in[3];
  const float* wd  = (const float*)d_in[4];
  float* out = (float*)d_out;

  // workspace carve-up (~65 MiB total), 256B aligned regions
  char* p = (char*)d_ws;
  auto carve = [&](size_t bytes) -> void* {
    void* r = (void*)p;
    p += (bytes + 255) & ~(size_t)255;
    return r;
  };
  __bf16* wgu_t = (__bf16*)carve((size_t)NEXP * 2 * FDIM * DDIM * 2); // 16.78 MB
  __bf16* wd_t  = (__bf16*)carve((size_t)NEXP * DDIM * FDIM * 2);     //  8.39 MB
  __bf16* xbf   = (__bf16*)carve((size_t)NTOK * DDIM * 2);            //  8.39 MB
  __bf16* H     = (__bf16*)carve((size_t)2 * NTOK * FDIM * 2);        // 33.55 MB
  int*    topIdx= (int*)  carve((size_t)2 * NTOK * 4);
  float*  topW  = (float*)carve((size_t)2 * NTOK * 4);
  int*    cnt   = (int*)  carve(NEXP * 4);
  int*    off   = (int*)  carve((NEXP + 1) * 4);
  int*    fill  = (int*)  carve(NEXP * 4);
  int*    btok  = (int*)  carve((size_t)2 * NTOK * 4);
  float*  bw    = (float*)carve((size_t)2 * NTOK * 4);

  // 0) zero output + counters
  zero_init<<<dim3((NTOK * DDIM) / 256), dim3(256), 0, stream>>>(out, cnt, fill);
  // 1) activations + weights -> bf16 (weights also K-contiguous transposed)
  cvt_x<<<dim3((NTOK * DDIM) / (256 * 4)), dim3(256), 0, stream>>>(x, xbf);
  transpose_cvt<<<dim3(2 * FDIM / 32, DDIM / 32, NEXP), dim3(32, 8), 0, stream>>>(
      wgu, wgu_t, DDIM, 2 * FDIM);
  transpose_cvt<<<dim3(DDIM / 32, FDIM / 32, NEXP), dim3(32, 8), 0, stream>>>(
      wd, wd_t, FDIM, DDIM);
  // 2-4) routing + compact dispatch
  router_topk<<<dim3(NTOK / 8), dim3(256), 0, stream>>>(x, rw, rb, topIdx, topW, cnt);
  scan8<<<dim3(1), dim3(32), 0, stream>>>(cnt, off);
  scatter_slots<<<dim3(2 * NTOK / 256), dim3(256), 0, stream>>>(topIdx, topW, off,
                                                                fill, btok, bw);
  // 5) gate/up GEMM + SwiGLU
  moe_gateup<<<dim3(FDIM / 64, NTOK / 64, NEXP), dim3(256), 0, stream>>>(
      xbf, wgu_t, btok, off, H);
  // 6) down GEMM + weighted scatter into out
  moe_down<<<dim3(DDIM / 128, NTOK / 64, NEXP), dim3(256), 0, stream>>>(
      H, wd_t, btok, bw, off, out);
}
